// PositionalEncoder_3882650437050
// MI455X (gfx1250) — matrix operand order repaired and checked by reading, not compiled, not run on hardware
//
#include <hip/hip_runtime.h>

typedef __attribute__((ext_vector_type(16))) _Float16 v16h;
typedef __attribute__((ext_vector_type(8)))  float    v8f;

#define NPTS    131072
#define IND     640          // 2*256 + 128
#define OUTD    512
#define MT      64           // rows per block
#define FSTRIDE 648          // feats LDS row stride in halves (640 + 8 pad -> bank-conflict free)
#define HSTRIDE 520          // h LDS row stride in halves (512 + 8 pad)
#define THREADS 256          // 8 wave32 waves

union Frag { v16h v; uint4 q[2]; };

static __device__ __forceinline__ v8f wmma_f16(const Frag& A, const Frag& B, v8f c) {
  return __builtin_amdgcn_wmma_f32_16x16x32_f16(false, A.v, false, B.v, (short)0, c, false, false);
}

// ---- prologue: f32 -> f16 weight conversion into workspace (stays hot in L2) ----
__global__ void convert_weights(const float* __restrict__ W1, const float* __restrict__ W2,
                                _Float16* __restrict__ w1h, _Float16* __restrict__ w2h) {
  const int n1 = IND * OUTD;
  const int n2 = OUTD * OUTD;
  const int stride = gridDim.x * blockDim.x;
  for (int k = blockIdx.x * blockDim.x + threadIdx.x; k < n1; k += stride) w1h[k] = (_Float16)W1[k];
  for (int k = blockIdx.x * blockDim.x + threadIdx.x; k < n2; k += stride) w2h[k] = (_Float16)W2[k];
}

// ---- fused: encode -> GEMM1 -> leakyReLU -> GEMM2, feats & h never leave LDS ----
__global__ __launch_bounds__(THREADS)
void fused_mlp(const float* __restrict__ pf, const float* __restrict__ pt,
               const float* __restrict__ nums,
               const float* __restrict__ Wsx, const float* __restrict__ bsx,
               const float* __restrict__ Wcx, const float* __restrict__ bcx,
               const float* __restrict__ Wsy, const float* __restrict__ bsy,
               const float* __restrict__ Wcy, const float* __restrict__ bcy,
               const float* __restrict__ Wn,  const float* __restrict__ bn,
               const _Float16* __restrict__ W1h, const float* __restrict__ b1,
               const _Float16* __restrict__ W2h, const float* __restrict__ b2,
               float* __restrict__ out) {
  extern __shared__ char smem_raw[];
  _Float16* feats = (_Float16*)smem_raw;              // MT x FSTRIDE halves
  _Float16* hbuf  = feats + MT * FSTRIDE;             // MT x HSTRIDE halves
  float*    sw    = (float*)(hbuf + MT * HSTRIDE);    // 768 enc weights + 320 row data

  const int tid  = threadIdx.x;
  const int lane = tid & 31;
  const int wave = tid >> 5;     // 0..7
  const int r16  = lane & 15;
  const int hi   = lane >> 4;
  const int row0 = blockIdx.x * MT;

  // stage tiny encoder weights + this block's point data in LDS
  if (tid < 64) {
    sw[tid]       = Wsx[tid];  sw[ 64 + tid] = bsx[tid];
    sw[128 + tid] = Wcx[tid];  sw[192 + tid] = bcx[tid];
    sw[256 + tid] = Wsy[tid];  sw[320 + tid] = bsy[tid];
    sw[384 + tid] = Wcy[tid];  sw[448 + tid] = bcy[tid];
  }
  if (tid < 128) { sw[512 + tid] = Wn[tid]; sw[640 + tid] = bn[tid]; }
  float* rowd = sw + 768;                             // xf|yf|xt|yt|num (64 each)
  if (tid < 64) {
    int row = row0 + tid;
    rowd[tid]       = pf[2 * row];
    rowd[ 64 + tid] = pf[2 * row + 1];
    rowd[128 + tid] = pt[2 * row];
    rowd[192 + tid] = pt[2 * row + 1];
    rowd[256 + tid] = nums[row];
  }
  __syncthreads();

  // ---- phase 1: positional encoding, f32 -> f16 into LDS ----
  // 8 rows per wave; each 32-lane column chunk lies inside one 64-wide segment
  // so the segment selection below is wave-uniform (no divergence, no division).
  for (int r = wave; r < MT; r += 8) {
    const float xf = rowd[r],       yf = rowd[ 64 + r];
    const float xt = rowd[128 + r], yt = rowd[192 + r];
    const float nm = rowd[256 + r];
    _Float16* frow = feats + r * FSTRIDE;
    #pragma unroll
    for (int cb = 0; cb < IND; cb += 32) {
      const int c = cb + lane;
      float v;
      if      (c <  64) { int j = c;       v = __sinf(xf * sw[j]       + sw[ 64 + j]); }
      else if (c < 128) { int j = c -  64; v = __cosf(xf * sw[128 + j] + sw[192 + j]); }
      else if (c < 192) { int j = c - 128; v = __sinf(yf * sw[256 + j] + sw[320 + j]); }
      else if (c < 256) { int j = c - 192; v = __cosf(yf * sw[384 + j] + sw[448 + j]); }
      else if (c < 320) { int j = c - 256; v = __sinf(xt * sw[j]       + sw[ 64 + j]); }
      else if (c < 384) { int j = c - 320; v = __cosf(xt * sw[128 + j] + sw[192 + j]); }
      else if (c < 448) { int j = c - 384; v = __sinf(yt * sw[256 + j] + sw[320 + j]); }
      else if (c < 512) { int j = c - 448; v = __cosf(yt * sw[384 + j] + sw[448 + j]); }
      else              { int j = c - 512; v = nm * sw[512 + j] + sw[640 + j]; }
      frow[c] = (_Float16)v;
    }
  }
  __syncthreads();

  // ---- phase 2: h = leakyReLU(feats @ W1^T + b1), kept as f16 in LDS ----
  // 2 n-tiles x 4 m-tiles of accumulators per wave: each A fragment read from
  // LDS feeds 2 WMMAs, each B fragment feeds 4 -> 8 WMMAs per 12 memory ops.
  for (int ntp = 0; ntp < 2; ++ntp) {
    const int nt0 = wave + ntp * 16;         // waves cover nt 0..31 in two passes
    const int nt1 = nt0 + 8;
    const int nc0 = nt0 * 16 + r16, nc1 = nt1 * 16 + r16;
    const _Float16* wB0 = W1h + (size_t)nc0 * IND + hi * 8;
    const _Float16* wB1 = W1h + (size_t)nc1 * IND + hi * 8;
    const float bc0 = b1[nc0], bc1 = b1[nc1];
    v8f acc0[4] = {{}, {}, {}, {}};
    v8f acc1[4] = {{}, {}, {}, {}};
    const _Float16* aB = feats + r16 * FSTRIDE + hi * 8;
    #pragma unroll 2
    for (int kb = 0; kb < IND / 32; ++kb) {
      Frag B0, B1;
      B0.q[0] = *(const uint4*)(wB0 + kb * 32);
      B0.q[1] = *(const uint4*)(wB0 + kb * 32 + 16);
      B1.q[0] = *(const uint4*)(wB1 + kb * 32);
      B1.q[1] = *(const uint4*)(wB1 + kb * 32 + 16);
      __builtin_prefetch(wB0 + (kb + 2) * 32, 0, 3);
      __builtin_prefetch(wB1 + (kb + 2) * 32, 0, 3);
      #pragma unroll
      for (int mt = 0; mt < 4; ++mt) {
        Frag A;
        const _Float16* a = aB + mt * 16 * FSTRIDE + kb * 32;
        A.q[0] = *(const uint4*)(a);
        A.q[1] = *(const uint4*)(a + 16);
        acc0[mt] = wmma_f16(A, B0, acc0[mt]);
        acc1[mt] = wmma_f16(A, B1, acc1[mt]);
      }
    }
    #pragma unroll
    for (int mt = 0; mt < 4; ++mt) {
      #pragma unroll
      for (int i = 0; i < 8; ++i) {         // C/D: VGPR i -> m = i + 8*hi, n = r16
        float h0 = acc0[mt][i] + bc0;
        float h1 = acc1[mt][i] + bc1;
        h0 = h0 >= 0.0f ? h0 : 0.01f * h0;
        h1 = h1 >= 0.0f ? h1 : 0.01f * h1;
        const int mrow = (mt * 16 + i + 8 * hi) * HSTRIDE + r16;
        hbuf[mrow + nt0 * 16] = (_Float16)h0;
        hbuf[mrow + nt1 * 16] = (_Float16)h1;
      }
    }
  }
  __syncthreads();

  // ---- phase 3: out = h @ W2^T + b2 (f32 straight to HBM, only HBM write) ----
  for (int ntp = 0; ntp < 2; ++ntp) {
    const int nt0 = wave + ntp * 16;
    const int nt1 = nt0 + 8;
    const int nc0 = nt0 * 16 + r16, nc1 = nt1 * 16 + r16;
    const _Float16* wB0 = W2h + (size_t)nc0 * OUTD + hi * 8;
    const _Float16* wB1 = W2h + (size_t)nc1 * OUTD + hi * 8;
    const float bc0 = b2[nc0], bc1 = b2[nc1];
    v8f acc0[4] = {{}, {}, {}, {}};
    v8f acc1[4] = {{}, {}, {}, {}};
    const _Float16* aB = hbuf + r16 * HSTRIDE + hi * 8;
    #pragma unroll 2
    for (int kb = 0; kb < OUTD / 32; ++kb) {
      Frag B0, B1;
      B0.q[0] = *(const uint4*)(wB0 + kb * 32);
      B0.q[1] = *(const uint4*)(wB0 + kb * 32 + 16);
      B1.q[0] = *(const uint4*)(wB1 + kb * 32);
      B1.q[1] = *(const uint4*)(wB1 + kb * 32 + 16);
      __builtin_prefetch(wB0 + (kb + 2) * 32, 0, 3);
      __builtin_prefetch(wB1 + (kb + 2) * 32, 0, 3);
      #pragma unroll
      for (int mt = 0; mt < 4; ++mt) {
        Frag A;
        const _Float16* a = aB + mt * 16 * HSTRIDE + kb * 32;
        A.q[0] = *(const uint4*)(a);
        A.q[1] = *(const uint4*)(a + 16);
        acc0[mt] = wmma_f16(A, B0, acc0[mt]);
        acc1[mt] = wmma_f16(A, B1, acc1[mt]);
      }
    }
    #pragma unroll
    for (int mt = 0; mt < 4; ++mt) {
      #pragma unroll
      for (int i = 0; i < 8; ++i) {
        const size_t obase = (size_t)(row0 + mt * 16 + i + 8 * hi) * OUTD + r16;
        out[obase + nt0 * 16] = acc0[mt][i] + bc0;
        out[obase + nt1 * 16] = acc1[mt][i] + bc1;
      }
    }
  }
}

extern "C" void kernel_launch(void* const* d_in, const int* in_sizes, int n_in,
                              void* d_out, int out_size, void* d_ws, size_t ws_size,
                              hipStream_t stream) {
  const float* pf   = (const float*)d_in[0];
  const float* pt   = (const float*)d_in[1];
  const float* nums = (const float*)d_in[2];
  const float* Wsx  = (const float*)d_in[3];
  const float* bsx  = (const float*)d_in[4];
  const float* Wcx  = (const float*)d_in[5];
  const float* bcx  = (const float*)d_in[6];
  const float* Wsy  = (const float*)d_in[7];
  const float* bsy  = (const float*)d_in[8];
  const float* Wcy  = (const float*)d_in[9];
  const float* bcy  = (const float*)d_in[10];
  const float* Wn   = (const float*)d_in[11];
  const float* bn   = (const float*)d_in[12];
  const float* W1   = (const float*)d_in[13];
  const float* b1   = (const float*)d_in[14];
  const float* W2   = (const float*)d_in[15];
  const float* b2   = (const float*)d_in[16];
  float* out = (float*)d_out;

  _Float16* w1h = (_Float16*)d_ws;
  _Float16* w2h = w1h + (size_t)IND * OUTD;

  const size_t smem = (size_t)(MT * FSTRIDE + MT * HSTRIDE) * sizeof(_Float16)
                    + (768 + 320) * sizeof(float);   // 153,856 B -> 2 workgroups / 320 KB WGP
  (void)hipFuncSetAttribute((const void*)fused_mlp,
                            hipFuncAttributeMaxDynamicSharedMemorySize, (int)smem);

  convert_weights<<<512, THREADS, 0, stream>>>(W1, W2, w1h, w2h);
  fused_mlp<<<NPTS / MT, THREADS, smem, stream>>>(pf, pt, nums,
                                                  Wsx, bsx, Wcx, bcx,
                                                  Wsy, bsy, Wcy, bcy,
                                                  Wn, bn, w1h, b1, w2h, b2, out);
}
